// PropagationBlock_15625091022908
// MI455X (gfx1250) — matrix operand order repaired and checked by reading, not compiled, not run on hardware
//
#include <hip/hip_runtime.h>

// CDNA5 / gfx1250 fused PropagationBlock kernel.
// Wave32; 8 waves/block; 16 edges per wave-tile; WMMA f16 16x16x32 for all matmuls.

typedef __attribute__((ext_vector_type(16))) _Float16 v16h;
typedef __attribute__((ext_vector_type(8)))  _Float16 v8h;
typedef __attribute__((ext_vector_type(8)))  float    v8f;

#define DDIM 32
#define FD   160            // 5*DDIM
#define WAVES 8
#define BLOCK (WAVES * 32)

__device__ __forceinline__ float fexp2(float x) { return __builtin_amdgcn_exp2f(x); }
__device__ __forceinline__ float frcp (float x) { return __builtin_amdgcn_rcpf(x); }
__device__ __forceinline__ float frsq (float x) { return __builtin_amdgcn_rsqf(x); }

// Hardware V_TANH_F32 on CDNA5 (TRANS32 op); guarded fallback to exp2 formula.
__device__ __forceinline__ float ftanh(float x) {
#if __has_builtin(__builtin_amdgcn_tanhf)
  return __builtin_amdgcn_tanhf(x);
#elif __has_builtin(__builtin_amdgcn_tanh_f32)
  return __builtin_amdgcn_tanh_f32(x);
#else
  return 1.0f - 2.0f * frcp(1.0f + fexp2(2.8853900817779268f * x));
#endif
}
// silu(x) = x / (1 + 2^(-x*log2(e)))
__device__ __forceinline__ float fsilu(float x) {
  return x * frcp(1.0f + fexp2(-1.4426950408889634f * x));
}
__device__ __forceinline__ void fadd_atomic(float* p, float v) {
  __hip_atomic_fetch_add(p, v, __ATOMIC_RELAXED, __HIP_MEMORY_SCOPE_AGENT);
}

// Wave-local LDS ordering: each wave uses a private stage region, so a full
// block barrier is unnecessary. DS ops complete in-order per wave; waiting for
// DScnt==0 makes all prior LDS stores visible to subsequent loads by any lane
// of the same wave. The "memory" clobber stops compiler reordering.
__device__ __forceinline__ void wave_lds_fence() {
  asm volatile("s_wait_dscnt 0" ::: "memory");
}

__global__ void zero_kernel(float* __restrict__ p, int n) {
  int i = blockIdx.x * blockDim.x + threadIdx.x;
  if (i < n) p[i] = 0.0f;
}

__global__ void __launch_bounds__(BLOCK, 1)
prop_kernel(const float* __restrict__ xn, const float* __restrict__ attr,
            const float* __restrict__ W1, const float* __restrict__ b1,
            const float* __restrict__ Wd1, const int* __restrict__ esrc,
            const int* __restrict__ edst, float* __restrict__ out, int nEdges) {
  // Wd1 in f16, pre-swizzled into WMMA B-fragment order:
  // element(frag=nt*5+kc, lane, i) = Wd1[kc*32 + 16*(i>>3) + 8*(lane>>4) + (i&7)][nt*16 + (lane&15)]
  __shared__ _Float16 sWd1[FD * FD];                 // 51200 B
  __shared__ _Float16 sStage[WAVES][16 * 32];        //  8192 B  (per-wave transpose buffer)

  const int tid  = threadIdx.x;
  const int lane = tid & 31;
  const int wave = tid >> 5;
  const int col  = lane & 15;   // N index (B/C/D) or A-row index
  const int hf   = lane >> 4;   // lane-half selects K subset / M subset

  // ---- one-time: swizzle Wd1 (f32 global -> f16 LDS fragments) ----
  for (int idx = tid; idx < FD * FD; idx += BLOCK) {
    int frag = idx >> 9;        // /512
    int rem  = idx & 511;
    int l    = rem >> 4;
    int i    = rem & 15;
    int nt   = frag / 5, kc = frag % 5;
    int n = nt * 16 + (l & 15);
    int k = kc * 32 + ((i >> 3) << 4) + ((l >> 4) << 3) + (i & 7);
    sWd1[idx] = (_Float16)Wd1[k * FD + n];
  }

  // ---- per-wave constants: W1 B-fragments (two 16-col tiles), 33rd row, bias ----
  v16h  w1frag[2];
  float w1last[2], biasv[2];
#pragma unroll
  for (int t = 0; t < 2; ++t) {
#pragma unroll
    for (int i = 0; i < 16; ++i)
      w1frag[t][i] = (_Float16)W1[(16 * (i >> 3) + 8 * hf + (i & 7)) * DDIM + 16 * t + col];
    w1last[t] = W1[32 * DDIM + 16 * t + col];
    biasv[t]  = b1[16 * t + col];
  }
  __syncthreads();   // sWd1 visible to all waves

  _Float16* stg = &sStage[wave][0];
  const int nBlockTiles = nEdges / (16 * WAVES);

  for (int bt = blockIdx.x; bt < nBlockTiles; bt += gridDim.x) {
    const int tile = bt * WAVES + wave;
    const int base = tile * 16;
    const int eRow = base + col;                     // this lane's A-matrix row (edge)

    // ================= fc1: W = silu(attr @ W1 + b1) via WMMA =================
    const float* arow = attr + (long)eRow * 33;
    v16h aAttr;
#pragma unroll
    for (int j = 0; j < 8; ++j) {
      aAttr[j]     = (_Float16)arow[8 * hf + j];
      aAttr[8 + j] = (_Float16)arow[16 + 8 * hf + j];
    }
    float a32[8];
#pragma unroll
    for (int r = 0; r < 8; ++r)
      a32[r] = attr[(long)(base + 8 * hf + r) * 33 + 32];

    v8f wcol[2];                                     // W in D (column) layout
#pragma unroll
    for (int t = 0; t < 2; ++t) {
      v8f c;
#pragma unroll
      for (int r = 0; r < 8; ++r) c[r] = biasv[t] + a32[r] * w1last[t];
      c = __builtin_amdgcn_wmma_f32_16x16x32_f16(false, aAttr, false, w1frag[t],
                                                 (short)0, c, false, false);
#pragma unroll
      for (int r = 0; r < 8; ++r) wcol[t][r] = fsilu(c[r]);
    }

    // transpose W (column layout -> row layout) through private LDS stage
#pragma unroll
    for (int t = 0; t < 2; ++t)
#pragma unroll
      for (int r = 0; r < 8; ++r)
        stg[(8 * hf + r) * 32 + 16 * t + col] = (_Float16)wcol[t][r];
    wave_lds_fence();
    float wrow[16];
    {
      v8h lo = *(const v8h*)&stg[col * 32 + 8 * hf];
      v8h hi = *(const v8h*)&stg[col * 32 + 16 + 8 * hf];
#pragma unroll
      for (int j = 0; j < 8; ++j) { wrow[j] = (float)lo[j]; wrow[8 + j] = (float)hi[j]; }
    }
    wave_lds_fence();

    // ================= gather + elementwise -> 5 A-fragments =================
    const int sIdx = esrc[eRow], dIdx = edst[eRow];
    float xsv[16], xdv[16];
    {
      const float4* ps0 = (const float4*)(xn + (long)sIdx * DDIM + 8 * hf);
      const float4* ps1 = (const float4*)(xn + (long)sIdx * DDIM + 16 + 8 * hf);
      const float4* pd0 = (const float4*)(xn + (long)dIdx * DDIM + 8 * hf);
      const float4* pd1 = (const float4*)(xn + (long)dIdx * DDIM + 16 + 8 * hf);
      float4 a = ps0[0], b = ps0[1], c = ps1[0], d = ps1[1];
      xsv[0]=a.x; xsv[1]=a.y; xsv[2]=a.z; xsv[3]=a.w;
      xsv[4]=b.x; xsv[5]=b.y; xsv[6]=b.z; xsv[7]=b.w;
      xsv[8]=c.x; xsv[9]=c.y; xsv[10]=c.z; xsv[11]=c.w;
      xsv[12]=d.x; xsv[13]=d.y; xsv[14]=d.z; xsv[15]=d.w;
      a = pd0[0]; b = pd0[1]; c = pd1[0]; d = pd1[1];
      xdv[0]=a.x; xdv[1]=a.y; xdv[2]=a.z; xdv[3]=a.w;
      xdv[4]=b.x; xdv[5]=b.y; xdv[6]=b.z; xdv[7]=b.w;
      xdv[8]=c.x; xdv[9]=c.y; xdv[10]=c.z; xdv[11]=c.w;
      xdv[12]=d.x; xdv[13]=d.y; xdv[14]=d.z; xdv[15]=d.w;
    }
    v16h Afr[5];
#pragma unroll
    for (int i = 0; i < 16; ++i) {
      float w = wrow[i];
      float g = w * (xsv[i] - xdv[i]);
      float a = w * (xsv[i] + xdv[i]) * 0.5f;
      Afr[0][i] = (_Float16)ftanh(g);
      Afr[1][i] = (_Float16)ftanh(a);
      Afr[2][i] = (_Float16)ftanh(g * a);
      Afr[3][i] = (_Float16)ftanh(g * g);
      Afr[4][i] = (_Float16)ftanh(a * a);
    }

    // ================= layer 1: x = tanh(dxe) @ Wd1 =================
    v8f acc[10];
#pragma unroll
    for (int nt = 0; nt < 10; ++nt) {
      v8f c = {0.f, 0.f, 0.f, 0.f, 0.f, 0.f, 0.f, 0.f};
#pragma unroll
      for (int kc = 0; kc < 5; ++kc) {
        const v8h* bp = (const v8h*)&sWd1[((nt * 5 + kc) * 32 + lane) * 16];
        v8h blo = bp[0], bhi = bp[1];
        v16h b;
#pragma unroll
        for (int j = 0; j < 8; ++j) { b[j] = blo[j]; b[8 + j] = bhi[j]; }
        c = __builtin_amdgcn_wmma_f32_16x16x32_f16(false, Afr[kc], false, b,
                                                   (short)0, c, false, false);
      }
      acc[nt] = c;
    }

    // ================= tv_norm (per edge row of 160) =================
    float s[8];
#pragma unroll
    for (int r = 0; r < 8; ++r) {
      float t = 0.f;
#pragma unroll
      for (int nt = 0; nt < 10; ++nt) t += acc[nt][r];
#pragma unroll
      for (int m = 1; m < 16; m <<= 1) t += __shfl_xor(t, m, 32);
      s[r] = t * (1.0f / 160.0f);
    }
#pragma unroll
    for (int nt = 0; nt < 10; ++nt)
#pragma unroll
      for (int r = 0; r < 8; ++r) acc[nt][r] -= s[r];
#pragma unroll
    for (int r = 0; r < 8; ++r) {
      float t = 0.f;
#pragma unroll
      for (int nt = 0; nt < 10; ++nt) t += acc[nt][r] * acc[nt][r];
#pragma unroll
      for (int m = 1; m < 16; m <<= 1) t += __shfl_xor(t, m, 32);
      s[r] = frsq(t + 1e-3f);
    }
#pragma unroll
    for (int nt = 0; nt < 10; ++nt)
#pragma unroll
      for (int r = 0; r < 8; ++r) acc[nt][r] *= s[r];

    // ====== layer 2: x = tanh(x) @ Wd1; transpose tanh(x) through stage ======
    v16h A2[5];
#pragma unroll
    for (int kc = 0; kc < 5; ++kc) {
#pragma unroll
      for (int p = 0; p < 2; ++p) {
        int nt = 2 * kc + p;
#pragma unroll
        for (int r = 0; r < 8; ++r)
          stg[(8 * hf + r) * 32 + 16 * p + col] = (_Float16)ftanh(acc[nt][r]);
      }
      wave_lds_fence();
      v8h lo = *(const v8h*)&stg[col * 32 + 8 * hf];
      v8h hi = *(const v8h*)&stg[col * 32 + 16 + 8 * hf];
#pragma unroll
      for (int j = 0; j < 8; ++j) { A2[kc][j] = lo[j]; A2[kc][8 + j] = hi[j]; }
      wave_lds_fence();
    }
#pragma unroll
    for (int nt = 0; nt < 10; ++nt) {
      v8f c = {0.f, 0.f, 0.f, 0.f, 0.f, 0.f, 0.f, 0.f};
#pragma unroll
      for (int kc = 0; kc < 5; ++kc) {
        const v8h* bp = (const v8h*)&sWd1[((nt * 5 + kc) * 32 + lane) * 16];
        v8h blo = bp[0], bhi = bp[1];
        v16h b;
#pragma unroll
        for (int j = 0; j < 8; ++j) { b[j] = blo[j]; b[8 + j] = bhi[j]; }
        c = __builtin_amdgcn_wmma_f32_16x16x32_f16(false, A2[kc], false, b,
                                                   (short)0, c, false, false);
      }
      acc[nt] = c;
    }

    // ====== msg = tile(W,5)*tanh(x); fold 5 blocks; scatter-add to out ======
    int sN[8], dN[8];
#pragma unroll
    for (int r = 0; r < 8; ++r) {
      sN[r] = esrc[base + 8 * hf + r];
      dN[r] = edst[base + 8 * hf + r];
    }
#pragma unroll
    for (int r = 0; r < 8; ++r) {
#pragma unroll
      for (int p = 0; p < 2; ++p) {
        float w  = wcol[p][r];
        float t0 = ftanh(acc[p][r]);                 // block 0 (gradX path)
        float S  = 0.f;
#pragma unroll
        for (int b = 1; b < 5; ++b) S += ftanh(acc[2 * b + p][r]);
        float sp = w * ( t0 + 0.5f * S);             // contribution to out[dst]
        float sm = w * (-t0 + 0.5f * S);             // contribution to out[src]
        int f = 16 * p + col;
        fadd_atomic(&out[(long)dN[r] * DDIM + f], sp);
        fadd_atomic(&out[(long)sN[r] * DDIM + f], sm);
      }
    }
  }
}

extern "C" void kernel_launch(void* const* d_in, const int* in_sizes, int n_in,
                              void* d_out, int out_size, void* d_ws, size_t ws_size,
                              hipStream_t stream) {
  const float* xn   = (const float*)d_in[0];
  const float* attr = (const float*)d_in[1];
  const float* W1   = (const float*)d_in[2];
  const float* b1   = (const float*)d_in[3];
  const float* Wd1  = (const float*)d_in[4];
  const int*   esrc = (const int*)d_in[5];
  const int*   edst = (const int*)d_in[6];
  float* out = (float*)d_out;
  const int nEdges = in_sizes[5];

  zero_kernel<<<(out_size + 255) / 256, 256, 0, stream>>>(out, out_size);

  int nbt  = nEdges / (16 * WAVES);   // 12500 block-tiles for E=1.6M
  int grid = nbt < 1250 ? nbt : 1250; // persistent blocks, amortize Wd1 LDS fill
  prop_kernel<<<grid, BLOCK, 0, stream>>>(xn, attr, W1, b1, Wd1, esrc, edst, out, nEdges);
}
